// YuKeMLPFLOW_37460704756277
// MI455X (gfx1250) — matrix operand order repaired and checked by reading, not compiled, not run on hardware
//
#include <hip/hip_runtime.h>
#include <math.h>

// ---------------------------------------------------------------------------
// MI455X (gfx1250) implementation of the per-latent MLP flow + JVP logdet.
// Memory-bound workload (136 MB of x @ 23.3 TB/s ~ 6us); compute routed
// through v_wmma_f32_16x16x32_f16 so the matrix units keep pace.
// ---------------------------------------------------------------------------

typedef __attribute__((ext_vector_type(16))) _Float16 v16h;
typedef __attribute__((ext_vector_type(8)))  float    v8f;

#define BATCH   8192
#define NLAT    64
#define HIDN    64
#define IN_DIM  65
#define SLOPE   0.2f
#define EPSV    1e-8f
#define WAVES   4
#define PAD     72   // padded f16 row stride: 144B -> conflict-free b128 LDS loads

__device__ __forceinline__ v8f wmma_f16(v16h a, v16h b, v8f c) {
    // (neg_a, A, neg_b, B, c_mod, C, reuse_a, reuse_b)
    return __builtin_amdgcn_wmma_f32_16x16x32_f16(false, a, false, b,
                                                  (short)0, c, false, false);
}

__device__ __forceinline__ v8f v8f_zero() {
    v8f z = {0.f, 0.f, 0.f, 0.f, 0.f, 0.f, 0.f, 0.f};
    return z;
}

__global__ __launch_bounds__(WAVES * 32)
void flow_kernel(const float* __restrict__ x,
                 const float* __restrict__ W0, const float* __restrict__ b0,
                 const float* __restrict__ W1, const float* __restrict__ b1,
                 const float* __restrict__ W2, const float* __restrict__ b2,
                 const float* __restrict__ W3, const float* __restrict__ b3,
                 float* __restrict__ resid, float* __restrict__ logws)
{
    // Weights for this latent, staged once per workgroup.
    __shared__ _Float16 sW0[HIDN][PAD];
    __shared__ _Float16 sW1[HIDN][PAD];
    __shared__ _Float16 sW2[HIDN][PAD];
    __shared__ float    sb[3][HIDN];
    __shared__ float    sw0c[HIDN];     // W0[:, 64] (tangent seed / rank-1 col)
    __shared__ float    sw3[HIDN];
    __shared__ float    sb3;
    // Per-wave private staging (C-layout -> A-layout transpose through LDS).
    __shared__ _Float16 sH[WAVES][2][16][PAD];   // [wave][primal|tangent][row][col]
    __shared__ float    sXl[WAVES][16];          // x[:, 64] per row
    __shared__ float    sP[WAVES][2][16][16];    // final-dot partials

    const int tid = threadIdx.x;
    const int l   = blockIdx.y;

    const float* W0g = W0 + (size_t)l * HIDN * IN_DIM;
    const float* W1g = W1 + (size_t)l * HIDN * HIDN;
    const float* W2g = W2 + (size_t)l * HIDN * HIDN;

    for (int i = tid; i < HIDN * HIDN; i += WAVES * 32) {
        const int j = i >> 6, k = i & 63;
        sW0[j][k] = (_Float16)W0g[j * IN_DIM + k];
        sW1[j][k] = (_Float16)W1g[i];
        sW2[j][k] = (_Float16)W2g[i];
    }
    if (tid < HIDN) {
        sw0c[tid]  = W0g[tid * IN_DIM + 64];
        sb[0][tid] = b0[l * HIDN + tid];
        sb[1][tid] = b1[l * HIDN + tid];
        sb[2][tid] = b2[l * HIDN + tid];
        sw3[tid]   = W3[l * HIDN + tid];
    }
    if (tid == 0) sb3 = b3[l];
    __syncthreads();

    const int wave  = tid >> 5;
    const int lane  = tid & 31;
    const int lhalf = lane & 15;       // row (A) / column (B,C)
    const int hi    = lane >> 4;       // K-half selector
    const int brow  = blockIdx.x * (WAVES * 16) + wave * 16;

    // x[b, l, :] row for this lane's A-fragment row.
    const float* xrow = x + ((size_t)(brow + lhalf) * NLAT + l) * IN_DIM;

    v8f cp[4], ct[4];
    #pragma unroll
    for (int t = 0; t < 4; ++t) cp[t] = v8f_zero();

    // ---------------- layer 0 (primal): z0 = x[:, :64] @ W0[:, :64]^T -------
    #pragma unroll
    for (int kb = 0; kb < 2; ++kb) {
        v16h a;
        const int koff = kb * 32 + hi * 16;  // A and B k order: contiguous k = 16h + s
        #pragma unroll
        for (int i = 0; i < 8; ++i) {
            a[i]     = (_Float16)xrow[koff + i];
            a[8 + i] = (_Float16)xrow[koff + 8 + i];
        }
        #pragma unroll
        for (int t = 0; t < 4; ++t) {
            v16h bm;
            const _Float16* wr = &sW0[t * 16 + lhalf][kb * 32 + hi * 16];
            #pragma unroll
            for (int i = 0; i < 16; ++i) bm[i] = wr[i];  // 2x ds_load_b128
            cp[t] = wmma_f16(a, bm, cp[t]);
        }
    }

    // Stage last input column (rank-1 update + tangent seed).
    if (lane < 16) sXl[wave][lane] = xrow[64];
    float xl[8];
    #pragma unroll
    for (int r = 0; r < 8; ++r) xl[r] = sXl[wave][r + hi * 8];

    // Layer-0 epilogue: bias + rank-1 + leaky; tangent dh0 = g * W0[:,64].
    #pragma unroll
    for (int t = 0; t < 4; ++t) {
        const int   col = t * 16 + lhalf;
        const float bj  = sb[0][col];
        const float wc  = sw0c[col];
        #pragma unroll
        for (int r = 0; r < 8; ++r) {
            const float z = cp[t][r] + bj + xl[r] * wc;
            const float g = (z > 0.0f) ? 1.0f : SLOPE;
            sH[wave][0][r + hi * 8][col] = (_Float16)(g * z);
            sH[wave][1][r + hi * 8][col] = (_Float16)(g * wc);
        }
    }

    float pp[8], pt[8];
    #pragma unroll
    for (int r = 0; r < 8; ++r) { pp[r] = 0.f; pt[r] = 0.f; }

    // ---------------- layers 1 & 2: primal + tangent share B-fragments ------
    #pragma unroll
    for (int layer = 1; layer <= 2; ++layer) {
        const _Float16 (*sW)[PAD] = (layer == 1) ? sW1 : sW2;

        v16h ap[2], at[2];
        #pragma unroll
        for (int kb = 0; kb < 2; ++kb) {
            const int koff = kb * 32 + hi * 16;
            #pragma unroll
            for (int i = 0; i < 8; ++i) {
                ap[kb][i]     = sH[wave][0][lhalf][koff + i];
                ap[kb][8 + i] = sH[wave][0][lhalf][koff + 8 + i];
                at[kb][i]     = sH[wave][1][lhalf][koff + i];
                at[kb][8 + i] = sH[wave][1][lhalf][koff + 8 + i];
            }
        }
        #pragma unroll
        for (int t = 0; t < 4; ++t) { cp[t] = v8f_zero(); ct[t] = v8f_zero(); }

        #pragma unroll
        for (int kb = 0; kb < 2; ++kb) {
            #pragma unroll
            for (int t = 0; t < 4; ++t) {
                v16h bm;
                const _Float16* wr = &sW[t * 16 + lhalf][kb * 32 + hi * 16];
                #pragma unroll
                for (int i = 0; i < 16; ++i) bm[i] = wr[i];
                cp[t] = wmma_f16(ap[kb], bm, cp[t]);
                ct[t] = wmma_f16(at[kb], bm, ct[t]);
            }
        }

        #pragma unroll
        for (int t = 0; t < 4; ++t) {
            const int   col = t * 16 + lhalf;
            const float bj  = sb[layer][col];
            const float w3v = sw3[col];
            #pragma unroll
            for (int r = 0; r < 8; ++r) {
                const float z  = cp[t][r] + bj;
                const float g  = (z > 0.0f) ? 1.0f : SLOPE;
                const float h  = g * z;
                const float dh = g * ct[t][r];
                if (layer == 2) {           // fuse final dot with W3
                    pp[r] += h  * w3v;
                    pt[r] += dh * w3v;
                } else {
                    sH[wave][0][r + hi * 8][col] = (_Float16)h;
                    sH[wave][1][r + hi * 8][col] = (_Float16)dh;
                }
            }
        }
    }

    // ---------------- layer 3: cross-lane reduce over 64 columns ------------
    #pragma unroll
    for (int r = 0; r < 8; ++r) {
        sP[wave][0][r + hi * 8][lhalf] = pp[r];
        sP[wave][1][r + hi * 8][lhalf] = pt[r];
    }
    float sum = 0.f;
    #pragma unroll
    for (int i = 0; i < 16; ++i) sum += sP[wave][hi][lhalf][i];

    if (hi == 0) {
        resid[(size_t)(brow + lhalf) * NLAT + l] = sum + sb3;
    } else {
        logws[(size_t)l * BATCH + (brow + lhalf)] = __logf(fabsf(sum) + EPSV);
    }
}

// Deterministic second pass: logabsdet[b] = sum_l log(|dJ|+eps).
__global__ __launch_bounds__(256)
void logdet_reduce(const float* __restrict__ logws, float* __restrict__ out)
{
    const int b = blockIdx.x * blockDim.x + threadIdx.x;
    if (b >= BATCH) return;
    float s = 0.f;
    #pragma unroll 8
    for (int l = 0; l < NLAT; ++l) s += logws[(size_t)l * BATCH + b];
    out[b] = s;
}

extern "C" void kernel_launch(void* const* d_in, const int* in_sizes, int n_in,
                              void* d_out, int out_size, void* d_ws, size_t ws_size,
                              hipStream_t stream)
{
    (void)in_sizes; (void)n_in; (void)out_size; (void)ws_size;
    const float* x  = (const float*)d_in[0];
    const float* W0 = (const float*)d_in[1];
    const float* b0 = (const float*)d_in[2];
    const float* W1 = (const float*)d_in[3];
    const float* b1 = (const float*)d_in[4];
    const float* W2 = (const float*)d_in[5];
    const float* b2 = (const float*)d_in[6];
    const float* W3 = (const float*)d_in[7];
    const float* b3 = (const float*)d_in[8];

    float* out   = (float*)d_out;
    float* logws = (float*)d_ws;     // BATCH*NLAT floats (2 MB) of scratch

    dim3 grid(BATCH / (WAVES * 16), NLAT);
    flow_kernel<<<grid, WAVES * 32, 0, stream>>>(x, W0, b0, W1, b1, W2, b2,
                                                 W3, b3, out, logws);
    logdet_reduce<<<BATCH / 256, 256, 0, stream>>>(logws,
                                                   out + (size_t)BATCH * NLAT);
}
